// BahdanauAttention_60327110639875
// MI455X (gfx1250) — compile-verified, matrix operand order checked by reading, not run on hardware
//
#include <hip/hip_runtime.h>
#include <hip/hip_bf16.h>
#include <stdint.h>

#define HIDDEN 1024
#define TWOH   2048
#define ATTN   512
#define BATCH  64
#define SEQ    2048

#define ROWS       32           // seq rows per workgroup in scores kernel
#define KCHUNK     512          // K elements staged per LDS chunk
#define LDS_STRIDE 520          // ushorts per row (520*2=1040B, multiple of 16, bank-skewed)

typedef __attribute__((ext_vector_type(16))) __bf16 bf16x16;
typedef __attribute__((ext_vector_type(8)))  float  f32x8;

union Frag16 { bf16x16 v; int4 q[2]; };

__device__ __forceinline__ unsigned int f32_to_bf16(float f) {
    unsigned int u = __float_as_uint(f);
    unsigned int r = u + 0x7FFFu + ((u >> 16) & 1u);   // round-to-nearest-even
    return r >> 16;
}

// ---------------------------------------------------------------------------
// K0: swizzle W_h (2048x512 fp32, row-major [k][n]) into bf16 WMMA B-fragment
// order: block (kt,nt) -> 32 lanes x 16 halfwords contiguous per lane.
// Lane L<16:  j=0..7 -> k=kt*32+j,      j=8..15 -> k=kt*32+16+(j-8)
// Lane L>=16: j=0..7 -> k=kt*32+8+j,    j=8..15 -> k=kt*32+24+(j-8)
// Column n = nt*16 + (L&15).
// ---------------------------------------------------------------------------
__global__ void wh_swizzle_kernel(const float* __restrict__ Wh,
                                  unsigned short* __restrict__ out) {
    const int nt = blockIdx.x;          // 0..31
    const int kt = blockIdx.y;          // 0..63
    const int L  = threadIdx.x;         // 0..31
    const int lhi = (L >= 16) ? 8 : 0;
    const int n   = nt * 16 + (L & 15);
    unsigned int pk[8];
#pragma unroll
    for (int p = 0; p < 8; ++p) {
        const int j0 = 2 * p, j1 = 2 * p + 1;
        const int k0 = kt * 32 + ((j0 < 8) ? j0 : j0 + 8) + lhi;
        const int k1 = kt * 32 + ((j1 < 8) ? j1 : j1 + 8) + lhi;
        const unsigned int lo = f32_to_bf16(Wh[(size_t)k0 * ATTN + n]);
        const unsigned int hi = f32_to_bf16(Wh[(size_t)k1 * ATTN + n]);
        pk[p] = lo | (hi << 16);
    }
    unsigned short* dst = out + ((size_t)(kt * 32 + nt) * 512 + L * 16);
    ((int4*)dst)[0] = make_int4((int)pk[0], (int)pk[1], (int)pk[2], (int)pk[3]);
    ((int4*)dst)[1] = make_int4((int)pk[4], (int)pk[5], (int)pk[6], (int)pk[7]);
}

// ---------------------------------------------------------------------------
// K1: dec_proj[b] = decoder_hidden[b] @ W_s   (64x1024 @ 1024x512), fp32.
// ---------------------------------------------------------------------------
__global__ void dec_proj_kernel(const float* __restrict__ dh,
                                const float* __restrict__ Ws,
                                float* __restrict__ dp) {
    __shared__ float s[HIDDEN];
    const int b = blockIdx.x, t = threadIdx.x;
#pragma unroll
    for (int j = 0; j < 4; ++j) s[t + j * 256] = dh[(size_t)b * HIDDEN + t + j * 256];
    __syncthreads();
    float a0 = 0.f, a1 = 0.f;
    for (int k = 0; k < HIDDEN; ++k) {
        const float x = s[k];
        a0 = fmaf(x, Ws[(size_t)k * ATTN + t],       a0);
        a1 = fmaf(x, Ws[(size_t)k * ATTN + t + 256], a1);
    }
    dp[(size_t)b * ATTN + t]       = a0;
    dp[(size_t)b * ATTN + t + 256] = a1;
}

// ---------------------------------------------------------------------------
// K2: fused scores kernel.
// scores[b,s] = sum_a v[a] * tanh(dec_proj[b,a] + (enc[b,s,:] @ W_h)[a])
// Grid (SEQ/32, BATCH), 256 threads = 8 waves.
// Wave w owns N-tiles w*4..w*4+3 and BOTH 16-row M-tiles of the 32-row tile:
// every B fragment is loaded once per WG and feeds 2 WMMAs; every A fragment
// feeds 4 WMMAs. Accumulators: 2x4 x (16x16 f32) = 64 VGPRs.
// ---------------------------------------------------------------------------
__global__ void scores_kernel(const float* __restrict__ enc,
                              const unsigned short* __restrict__ whswz,
                              const float* __restrict__ dp,
                              const float* __restrict__ v,
                              float* __restrict__ scores) {
    __shared__ unsigned short sA[ROWS * LDS_STRIDE];   // 33,280 B bf16 A tile
    __shared__ float sScore[ROWS];

    const int b     = blockIdx.y;
    const int s0    = blockIdx.x * ROWS;
    const int t     = threadIdx.x;
    const int lane  = t & 31;
    const int wave  = t >> 5;          // owns N-tiles wave*4 .. wave*4+3
    const int l16   = lane & 15;
    const int lhalf = lane >> 4;

    if (t < ROWS) sScore[t] = 0.f;

    f32x8 acc[2][4] = {};              // [m-tile][n-tile] accumulators

    const int ldrow = t >> 3;          // staging: 8 threads per row
    const int qbase = t & 7;           // float4 slot within row
    const float* encRow = enc + ((size_t)b * SEQ + s0 + ldrow) * TWOH;
    unsigned short* dstRow = &sA[ldrow * LDS_STRIDE];

    for (int kc = 0; kc < TWOH / KCHUNK; ++kc) {
        // --- stage 32 x 512 fp32 -> bf16 LDS tile (coalesced float4 loads) ---
        const float* src = encRow + kc * KCHUNK;
#pragma unroll
        for (int u = 0; u < 16; ++u) {
            const int q = qbase + u * 8;               // float4 index 0..127
            const float4 f = ((const float4*)src)[q];
            const unsigned int p0 = f32_to_bf16(f.x) | (f32_to_bf16(f.y) << 16);
            const unsigned int p1 = f32_to_bf16(f.z) | (f32_to_bf16(f.w) << 16);
            *(uint2*)&dstRow[q * 4] = make_uint2(p0, p1);
        }
        if (kc < TWOH / KCHUNK - 1) {
            // pull next chunk toward L2 while we compute (global_prefetch_b8)
            __builtin_prefetch(src + KCHUNK + qbase * 64, 0, 0);
            __builtin_prefetch(src + KCHUNK + qbase * 64 + 32, 0, 0);
        }
        __syncthreads();

        // --- 16 k-steps of 32 over this chunk ---
        const int ktbase = kc * (KCHUNK / 32);
#pragma unroll 1
        for (int ks = 0; ks < KCHUNK / 32; ++ks) {
            Frag16 af[2];
#pragma unroll
            for (int mt = 0; mt < 2; ++mt) {
                const unsigned short* ap =
                    &sA[(mt * 16 + l16) * LDS_STRIDE + ks * 32 + lhalf * 8];
                af[mt].q[0] = ((const int4*)ap)[0];        // K +0..7   (or +8..15)
                af[mt].q[1] = ((const int4*)(ap + 16))[0]; // K +16..23 (or +24..31)
            }
            const unsigned short* wb =
                whswz + (((size_t)(ktbase + ks) * 32 + wave * 4) * 512) + lane * 16;
#pragma unroll
            for (int nt = 0; nt < 4; ++nt) {
                Frag16 bf_;
                const unsigned short* p = wb + (size_t)nt * 512;
                bf_.q[0] = ((const int4*)p)[0];
                bf_.q[1] = ((const int4*)p)[1];
#pragma unroll
                for (int mt = 0; mt < 2; ++mt) {
                    acc[mt][nt] = __builtin_amdgcn_wmma_f32_16x16x32_bf16(
                        false, af[mt].v, false, bf_.v, (short)0, acc[mt][nt],
                        false, false);
                }
            }
        }
        __syncthreads();
    }

    // --- fused epilogue: + dec_proj, tanh, dot with v, reduce over A ---
    float dpv[4], vv[4];
#pragma unroll
    for (int nt = 0; nt < 4; ++nt) {
        const int col = (wave * 4 + nt) * 16 + l16;
        dpv[nt] = dp[(size_t)b * ATTN + col];
        vv[nt]  = v[col];
    }
#pragma unroll
    for (int mt = 0; mt < 2; ++mt) {
#pragma unroll
        for (int i = 0; i < 8; ++i) {
            float part = 0.f;
#pragma unroll
            for (int nt = 0; nt < 4; ++nt)
                part = fmaf(tanhf(acc[mt][nt][i] + dpv[nt]), vv[nt], part);
            // reduce across the 16 lanes of this half-wave (same output row)
#pragma unroll
            for (int off = 8; off >= 1; off >>= 1)
                part += __shfl_xor(part, off, 16);
            if (l16 == 0) {
                const int row = mt * 16 + lhalf * 8 + i;
                atomicAdd(&sScore[row], part);             // ds_add_f32
            }
        }
    }
    __syncthreads();
    if (t < ROWS) scores[(size_t)b * SEQ + s0 + t] = sScore[t];
}

// ---------------------------------------------------------------------------
// K3: masked softmax over S per batch row; writes weights to d_out region.
// ---------------------------------------------------------------------------
__global__ void softmax_kernel(const float* __restrict__ scores,
                               const unsigned char* __restrict__ mask,
                               float* __restrict__ weights) {
    __shared__ float red[256];
    const int b = blockIdx.x, t = threadIdx.x;
    float x[8];
#pragma unroll
    for (int j = 0; j < 8; ++j) {
        const int s = t + j * 256;
        const float sc = scores[(size_t)b * SEQ + s];
        x[j] = mask[(size_t)b * SEQ + s] ? sc : -INFINITY;
    }
    float m = x[0];
#pragma unroll
    for (int j = 1; j < 8; ++j) m = fmaxf(m, x[j]);
    red[t] = m;
    __syncthreads();
    for (int off = 128; off > 0; off >>= 1) {
        if (t < off) red[t] = fmaxf(red[t], red[t + off]);
        __syncthreads();
    }
    m = red[0];
    __syncthreads();
    float sum = 0.f;
#pragma unroll
    for (int j = 0; j < 8; ++j) { x[j] = __expf(x[j] - m); sum += x[j]; }
    red[t] = sum;
    __syncthreads();
    for (int off = 128; off > 0; off >>= 1) {
        if (t < off) red[t] += red[t + off];
        __syncthreads();
    }
    const float inv = 1.0f / red[0];
#pragma unroll
    for (int j = 0; j < 8; ++j)
        weights[(size_t)b * SEQ + t + j * 256] = x[j] * inv;
}

// ---------------------------------------------------------------------------
// K4: context[b,h] = sum_s w[b,s] * enc[b,s,h]. Streaming float4 pass.
// Grid (TWOH/1024, BATCH), 256 threads; each thread owns 4 columns.
// ---------------------------------------------------------------------------
__global__ void context_kernel(const float* __restrict__ enc,
                               const float* __restrict__ weights,
                               float* __restrict__ ctx) {
    __shared__ float sw[SEQ];
    const int b = blockIdx.y;
    const int cbase = blockIdx.x * 1024 + threadIdx.x * 4;
#pragma unroll
    for (int j = 0; j < 8; ++j)
        sw[threadIdx.x + j * 256] = weights[(size_t)b * SEQ + threadIdx.x + j * 256];
    __syncthreads();
    float4 acc = make_float4(0.f, 0.f, 0.f, 0.f);
    const float* base = enc + (size_t)b * SEQ * TWOH + cbase;
    for (int s = 0; s < SEQ; ++s) {
        const float w = sw[s];
        const float4 e = *(const float4*)(base + (size_t)s * TWOH);
        acc.x = fmaf(w, e.x, acc.x);
        acc.y = fmaf(w, e.y, acc.y);
        acc.z = fmaf(w, e.z, acc.z);
        acc.w = fmaf(w, e.w, acc.w);
    }
    *(float4*)(ctx + (size_t)b * TWOH + cbase) = acc;
}

// ---------------------------------------------------------------------------
extern "C" void kernel_launch(void* const* d_in, const int* in_sizes, int n_in,
                              void* d_out, int out_size, void* d_ws, size_t ws_size,
                              hipStream_t stream) {
    (void)in_sizes; (void)n_in; (void)out_size; (void)ws_size;
    const float*         dh   = (const float*)d_in[0];          // (64,1024)
    const float*         enc  = (const float*)d_in[1];          // (64,2048,2048)
    const unsigned char* mask = (const unsigned char*)d_in[2];  // (64,2048) bool
    const float*         Ws   = (const float*)d_in[3];          // (1024,512)
    const float*         Wh   = (const float*)d_in[4];          // (2048,512)
    const float*         v    = (const float*)d_in[5];          // (512,)

    char* ws = (char*)d_ws;
    unsigned short* whswz  = (unsigned short*)ws;                         // 2 MiB
    float*          dp     = (float*)(ws + (size_t)TWOH * ATTN * 2);      // 128 KiB
    float*          scores = (float*)(ws + (size_t)TWOH * ATTN * 2
                                         + (size_t)BATCH * ATTN * 4);     // 512 KiB

    float* ctx = (float*)d_out;                        // (64,2048) context
    float* wts = (float*)d_out + (size_t)BATCH * SEQ;  // (64,2048) attention weights

    wh_swizzle_kernel<<<dim3(ATTN / 16, TWOH / 32), 32, 0, stream>>>(Wh, whswz);
    dec_proj_kernel  <<<BATCH, 256, 0, stream>>>(dh, Ws, dp);
    scores_kernel    <<<dim3(SEQ / ROWS, BATCH), 256, 0, stream>>>(enc, whswz, dp, v, scores);
    softmax_kernel   <<<BATCH, 256, 0, stream>>>(scores, mask, wts);
    context_kernel   <<<dim3(TWOH / 1024, BATCH), 256, 0, stream>>>(enc, wts, ctx);
}
